// BoundaryKDV7_24979529793879
// MI455X (gfx1250) — compile-verified, matrix operand order checked by reading, not run on hardware
//
#include <hip/hip_runtime.h>

#define NCLS   14
#define KCLS   13          // boundary classes 1..13
#define HDIM   512
#define WDIM   512
#define PPX    (HDIM * WDIM)   // 262144 pixels per image
#define BATCH  8
#define NBUCKET (BATCH * KCLS) // 104

typedef __attribute__((ext_vector_type(16))) _Float16 v16h;
typedef __attribute__((ext_vector_type(8)))  float    v8f;

// ---------------------------------------------------------------------------
// ws layout (dwords): [0..103] f32 kl_sum, [104..207] u32 n, [208..311] u32 npos
// ---------------------------------------------------------------------------
__global__ void bkd_zero(unsigned int* __restrict__ ws) {
    int i = threadIdx.x;
    if (i < 3 * NBUCKET) ws[i] = 0u;
}

// ---------------------------------------------------------------------------
// Streaming kernel: 1 thread = 1 pixel. Memory-bound (~235 MB preds once).
// ---------------------------------------------------------------------------
__global__ __launch_bounds__(256) void bkd_main(
        const float* __restrict__ S, const float* __restrict__ T,
        const int*   __restrict__ G,
        float* __restrict__ ws_kl, unsigned int* __restrict__ ws_n,
        unsigned int* __restrict__ ws_np) {
    __shared__ float        s_kl[KCLS];
    __shared__ unsigned int s_n [KCLS];
    __shared__ unsigned int s_np[KCLS];

    const int tid = threadIdx.x;
    if (tid < KCLS) { s_kl[tid] = 0.0f; s_n[tid] = 0u; s_np[tid] = 0u; }
    __syncthreads();

    const int b     = blockIdx.y;
    const int pix   = blockIdx.x * 256 + tid;
    const int h     = pix >> 9;
    const int w     = pix & (WDIM - 1);
    const int gbase = b * PPX;
    const int g     = G[gbase + pix];

    // boundary = in-mask AND (at border OR any 4-neighbor differs)
    bool bnd = false;
    if (g >= 1) {
        if (h == 0 || h == HDIM - 1 || w == 0 || w == WDIM - 1) {
            bnd = true;
        } else {
            bnd = (G[gbase + pix - WDIM] != g) | (G[gbase + pix + WDIM] != g) |
                  (G[gbase + pix - 1]    != g) | (G[gbase + pix + 1]    != g);
        }
    }

    if (bnd) {
        float sv[NCLS], tv[NCLS];
        const int base = b * NCLS * PPX + pix;     // channel stride = PPX
#pragma unroll
        for (int c = 0; c < NCLS; ++c) {
            sv[c] = S[base + c * PPX];             // coalesced across lanes
            tv[c] = T[base + c * PPX];
        }
        float mS = sv[0], mT = tv[0];
#pragma unroll
        for (int c = 1; c < NCLS; ++c) {
            mS = fmaxf(mS, sv[c]);
            mT = fmaxf(mT, tv[c]);
        }
        float seS = 0.0f, seT = 0.0f, wacc = 0.0f;
#pragma unroll
        for (int c = 0; c < NCLS; ++c) {
            float dS = sv[c] - mS;
            float dT = tv[c] - mT;
            seS += expf(dS);
            float eT = expf(dT);
            seT += eT;
            wacc = fmaf(eT, dT - dS, wacc);        // sum eT*(dT - dS)
        }
        // KL_pixel = sum_c pT*(log pT - log pS)
        //          = wacc/seT + log(seS) - log(seT)
        float kl = wacc / seT + logf(seS) - logf(seT);

        const int k = g - 1;
        atomicAdd(&s_kl[k], kl);                   // ds_add_f32
        atomicAdd(&s_n[k], 1u);
        if (pix > 0) atomicAdd(&s_np[k], 1u);      // exact idx_sum>0 test
    }
    __syncthreads();

    if (tid < KCLS && s_n[tid]) {
        const int o = b * KCLS + tid;
        atomicAdd(&ws_kl[o], s_kl[tid]);
        atomicAdd(&ws_n [o], s_n [tid]);
        atomicAdd(&ws_np[o], s_np[tid]);
    }
}

// ---------------------------------------------------------------------------
// Finisher: one wave. Per-lane partial over the 104 buckets, then a wave-wide
// sum via v_wmma_f32_16x16x32_f16 with an all-ones A matrix (layout-agnostic:
// D[m,n] = sum_k B[k,n], f32 accumulate). Each partial is split into exact
// f16 hi+lo pair so the WMMA sum carries f32-grade precision.
// ---------------------------------------------------------------------------
__global__ void bkd_final(const float* __restrict__ ws_kl,
                          const unsigned int* __restrict__ ws_n,
                          const unsigned int* __restrict__ ws_np,
                          float* __restrict__ out) {
    const int lane = threadIdx.x;                  // 32 threads = 1 wave
    float partial = 0.0f;
    for (int i = lane; i < NBUCKET; i += 32) {
        if (ws_np[i] > 0u) {
            unsigned int nn = ws_n[i];
            float n = (float)(nn < 1u ? 1u : nn);
            partial += ws_kl[i] / (14.0f * n);     // TAU^2 = LOSS_WEIGHT = 1
        }
    }

    _Float16 hi = (_Float16)partial;
    _Float16 lo = (_Float16)(partial - (float)hi);

    v16h a, bm;
#pragma unroll
    for (int j = 0; j < 16; ++j) { a[j] = (_Float16)1.0f; bm[j] = (_Float16)0.0f; }
    bm[0] = hi;
    bm[1] = lo;
    v8f c = {};
    // D[m,n] = sum_k 1 * B[k,n]; total of all B cells = sum of lane partials
    v8f d = __builtin_amdgcn_wmma_f32_16x16x32_f16(
        /*neg_a=*/false, a, /*neg_b=*/false, bm,
        /*c_mod=*/(short)0, c, /*reuse_a=*/false, /*reuse_b=*/false);

    __shared__ float red[16];
    if (lane < 16) red[lane] = d[0];               // D row 0: N = lane
    __syncthreads();
    if (lane == 0) {
        float t = 0.0f;
#pragma unroll
        for (int j = 0; j < 16; ++j) t += red[j];
        out[0] = t;
    }
}

// ---------------------------------------------------------------------------
extern "C" void kernel_launch(void* const* d_in, const int* in_sizes, int n_in,
                              void* d_out, int out_size, void* d_ws, size_t ws_size,
                              hipStream_t stream) {
    (void)in_sizes; (void)n_in; (void)out_size; (void)ws_size;
    const float* S = (const float*)d_in[0];
    const float* T = (const float*)d_in[1];
    const int*   G = (const int*)d_in[2];
    float* out = (float*)d_out;

    float*        ws_kl = (float*)d_ws;
    unsigned int* ws_n  = (unsigned int*)d_ws + NBUCKET;
    unsigned int* ws_np = (unsigned int*)d_ws + 2 * NBUCKET;

    bkd_zero<<<1, 384, 0, stream>>>((unsigned int*)d_ws);
    dim3 grid(PPX / 256, BATCH);
    bkd_main<<<grid, 256, 0, stream>>>(S, T, G, ws_kl, ws_n, ws_np);
    bkd_final<<<1, 32, 0, stream>>>(ws_kl, ws_n, ws_np, out);
}
